// AssociativeAttention_14104672600706
// MI455X (gfx1250) — compile-verified, matrix operand order checked by reading, not run on hardware
//
#include <hip/hip_runtime.h>
#include <hip/hip_bf16.h>

// ---------------- CDNA5 WMMA plumbing (gfx1250, wave32) ----------------
typedef __attribute__((ext_vector_type(16))) __bf16 v16bf;
typedef __attribute__((ext_vector_type(8)))  float  v8f;
typedef __attribute__((ext_vector_type(4)))  unsigned int u32x4;

struct FragU  { u32x4 lo, hi; };           // 32B = 16 bf16
static_assert(sizeof(FragU) == 32, "frag size");

#define BB 2
#define LL 1024
#define DD 512
#define HH 8
#define hh 64
#define KK 24
#define EPSF 1e-5f

// Async global->LDS staging (CDNA5 GLOBAL_LOAD_ASYNC_TO_LDS_B128, ASYNCcnt).
#define USE_ASYNC_LDS 1

__device__ __forceinline__ unsigned short f2bf(float f) {
    unsigned u = __float_as_uint(f);
    unsigned r = u + 0x7FFFu + ((u >> 16) & 1u);   // round-to-nearest-even
    return (unsigned short)(r >> 16);
}

#if USE_ASYNC_LDS
// Per-lane 16B copy global -> LDS, tracked by ASYNCcnt (ISA 15.18.3 op 98).
__device__ __forceinline__ void async_g2l_b128(unsigned lds_byte_addr, const void* gaddr) {
    asm volatile("global_load_async_to_lds_b128 %0, %1, off"
                 :: "v"(lds_byte_addr), "v"((unsigned long long)gaddr)
                 : "memory");
}
__device__ __forceinline__ void wait_asynccnt0() {
    asm volatile("s_wait_asynccnt 0" ::: "memory");
}
#endif

// 16x32 bf16 fragment, source row-major, rows contiguous at stride ld.
// base points at [row0, k0]. Lane t: row = t&15; K halves per ISA 16-bit A/B layout.
__device__ __forceinline__ v16bf load_frag_rm(const unsigned short* base, int ld, int lane) {
    int r = lane & 15, half = lane >> 4;
    const unsigned short* p = base + (long)r * ld + half * 8;
    FragU f;
    f.lo = *(const u32x4*)(p);
    f.hi = *(const u32x4*)(p + 16);
    return __builtin_bit_cast(v16bf, f);
}

// Same, but row index clamped (for tail tiles past L).
__device__ __forceinline__ v16bf load_frag_rm_clamp(const unsigned short* src, int ld,
                                                    int row0, int maxRow, int colOff, int lane) {
    int r = row0 + (lane & 15); if (r > maxRow) r = maxRow;
    int half = lane >> 4;
    const unsigned short* p = src + (long)r * ld + colOff + half * 8;
    FragU f;
    f.lo = *(const u32x4*)(p);
    f.hi = *(const u32x4*)(p + 16);
    return __builtin_bit_cast(v16bf, f);
}

__device__ __forceinline__ v8f wmma_bf16(v16bf a, v16bf b, v8f c) {
    return __builtin_amdgcn_wmma_f32_16x16x32_bf16(false, a, false, b, (short)0, c, false, false);
}

// ---------------- K0: conversions ----------------
__global__ void cvt_f32_to_bf16(const float* __restrict__ x, unsigned short* __restrict__ o, int n) {
    for (int i = blockIdx.x * blockDim.x + threadIdx.x; i < n; i += gridDim.x * blockDim.x)
        o[i] = f2bf(x[i]);
}

// Wt[n*512 + k] = bf16(W[k*512 + n])  (column-major weight for contiguous-K fragments)
__global__ void transpose_w_bf16(const float* __restrict__ W, unsigned short* __restrict__ Wt) {
    int idx = blockIdx.x * blockDim.x + threadIdx.x;
    if (idx >= DD * DD) return;
    int n = idx >> 9, k = idx & (DD - 1);
    Wt[idx] = f2bf(W[(long)k * DD + n]);
}

// ---------------- K1: f_proj = spectral_filters @ Wtd + btd  (L x D) ----------------
__global__ void fproj_kernel(const float* __restrict__ sf, const float* __restrict__ Wtd,
                             const float* __restrict__ btd, float* __restrict__ fp) {
    int idx = blockIdx.x * blockDim.x + threadIdx.x;
    if (idx >= LL * DD) return;
    int l = idx >> 9, c = idx & (DD - 1);
    float a = btd[c];
#pragma unroll
    for (int j = 0; j < KK; ++j) a += sf[l * KK + j] * Wtd[j * DD + c];
    fp[idx] = a;
}

// ---------------- K2: bf16 WMMA GEMM  C[M,N] = A[M,512] @ Wt[N,512]^T + bias ----------------
// grid = (M/16, N/64), block = 128 (4 waves, one 16x16 n-tile each).
// A row-tile (16x512 bf16 = 16KB) staged once into LDS via async copies and shared
// by all 4 waves (removes 4x-redundant global A traffic).
__global__ void gemm_bf16_kernel(const unsigned short* __restrict__ A,
                                 const unsigned short* __restrict__ Wt,
                                 const float* __restrict__ bias,
                                 float* __restrict__ Cf, unsigned short* __restrict__ Cbf,
                                 int M, int N) {
    int wave = threadIdx.x >> 5, lane = threadIdx.x & 31;
    int m0 = blockIdx.x * 16;
    int n0 = blockIdx.y * 64 + wave * 16;
    const unsigned short* Ab = A  + (long)m0 * DD;
    const unsigned short* Bb = Wt + (long)n0 * DD;

#if USE_ASYNC_LDS
    __shared__ __align__(16) unsigned short tileA[16 * DD];   // 16KB
    {
        int t = threadIdx.x;                  // 128 threads * 8 passes * 8 elem = 16*512
#pragma unroll
        for (int p = 0; p < 8; ++p) {
            int off = p * 1024 + t * 8;       // bf16 elements
            async_g2l_b128((unsigned)(unsigned long long)(const void*)&tileA[off],
                           (const void*)(Ab + off));
        }
        wait_asynccnt0();
        __syncthreads();
    }
#endif

    v8f acc = {0.f, 0.f, 0.f, 0.f, 0.f, 0.f, 0.f, 0.f};
    for (int k0 = 0; k0 < DD; k0 += 32) {
        if (k0 + 64 < DD) __builtin_prefetch((const void*)(Bb + k0 + 64), 0, 0);
#if USE_ASYNC_LDS
        v16bf a = load_frag_rm(&tileA[k0], DD, lane);   // ds_load_b128
#else
        v16bf a = load_frag_rm(Ab + k0, DD, lane);
#endif
        v16bf b = load_frag_rm(Bb + k0, DD, lane);
        acc = wmma_bf16(a, b, acc);
    }
    int half = lane >> 4, col = n0 + (lane & 15);
    float bc = bias ? bias[col] : 0.f;
#pragma unroll
    for (int r = 0; r < 8; ++r) {
        int row = m0 + r + half * 8;
        float v = acc[r] + bc;
        if (Cf)  Cf[(long)row * N + col] = v;
        if (Cbf) Cbf[(long)row * N + col] = f2bf(v);
    }
}

// ---------------- K3: sim + in-place L2 normalize of k,v ----------------
// grid = B*L, block = 512 (one thread per D column; 64-thread groups per head)
__global__ void simnorm_kernel(const float* __restrict__ q, float* __restrict__ k, float* __restrict__ v,
                               const float* __restrict__ qk_scale, float* __restrict__ sim) {
    int bl = blockIdx.x, tid = threadIdx.x;
    int hd = tid >> 6, d = tid & 63;
    long base = (long)bl * DD + tid;
    float qv = q[base], kv = k[base], vv = v[base];
    __shared__ float red[DD];

    float sqk, skk, svv;
    red[tid] = qv * kv; __syncthreads();
    for (int off = 32; off >= 1; off >>= 1) { if (d < off) red[tid] += red[tid + off]; __syncthreads(); }
    sqk = red[hd * 64]; __syncthreads();

    red[tid] = kv * kv; __syncthreads();
    for (int off = 32; off >= 1; off >>= 1) { if (d < off) red[tid] += red[tid + off]; __syncthreads(); }
    skk = red[hd * 64]; __syncthreads();

    red[tid] = vv * vv; __syncthreads();
    for (int off = 32; off >= 1; off >>= 1) { if (d < off) red[tid] += red[tid + off]; __syncthreads(); }
    svv = red[hd * 64];

    if (d == 0) {
        int b = bl >> 10, l = bl & (LL - 1);
        sim[((long)b * HH + hd) * LL + l] = sqk * qk_scale[hd];
    }
    k[base] = kv / fmaxf(sqrtf(skk), 1e-12f);
    v[base] = vv / fmaxf(sqrtf(svv), 1e-12f);
}

// ---------------- K4: causal conv  k_t[l,d] = sum_{m<=l} f[l-m,d] * k_norm[m,d] ----------------
// grid = (B*H, L/4), block = (64,4).
// Emits: kt/vt f32 (gates), vtbf row-major bf16 (attn 1st GEMM B),
//        ktT transposed bf16 [ (b*H+hd)*64 + e ][ m ] (attn 2nd GEMM B, contraction-contiguous).
__global__ void conv_kernel(const float* __restrict__ kn, const float* __restrict__ vn,
                            const float* __restrict__ fp,
                            float* __restrict__ kt, float* __restrict__ vt,
                            unsigned short* __restrict__ ktT, unsigned short* __restrict__ vtbf) {
    int bh = blockIdx.x, b = bh >> 3, hd = bh & 7;
    int l = blockIdx.y * 4 + threadIdx.y;
    int d = threadIdx.x;
    int col = hd * 64 + d;
    long rowb = (long)b * LL;
    float ak = 0.f, av = 0.f;
    for (int m = 0; m <= l; ++m) {
        float f = fp[(long)(l - m) * DD + col];
        ak = fmaf(f, kn[(rowb + m) * DD + col], ak);
        av = fmaf(f, vn[(rowb + m) * DD + col], av);
    }
    long o = (rowb + l) * DD + col;
    kt[o] = ak; vt[o] = av;
    vtbf[o] = f2bf(av);
    ktT[((long)bh * 64 + d) * LL + l] = f2bf(ak);
}

// ---------------- K5: gates[b,h,l] = relu(v_t^T (scale*Wg) k_t + bg)^2 + EPS ----------------
// grid = (B*H, L/64), block = 64
__global__ void gate_kernel(const float* __restrict__ kt, const float* __restrict__ vt,
                            const float* __restrict__ Wg, const float* __restrict__ bg,
                            const float* __restrict__ kv_scale, float* __restrict__ gates) {
    int bh = blockIdx.x, b = bh >> 3, hd = bh & 7;
    int d = threadIdx.x;
    __shared__ float red[64];
    __shared__ float krow[64];
    for (int li = 0; li < 64; ++li) {
        int l = blockIdx.y * 64 + li;
        long o = ((long)(b * LL + l)) * DD + hd * 64;
        krow[d] = kt[o + d];
        __syncthreads();
        float t = 0.f;
#pragma unroll 8
        for (int e = 0; e < 64; ++e)
            t = fmaf(kv_scale[hd * 4096 + d * 64 + e] * Wg[d * 64 + e], krow[e], t);
        red[d] = vt[o + d] * t;
        __syncthreads();
        for (int off = 32; off >= 1; off >>= 1) { if (d < off) red[d] += red[d + off]; __syncthreads(); }
        if (d == 0) {
            float lg = red[0] + bg[0];
            float rl = lg > 0.f ? lg : 0.f;
            gates[(long)bh * LL + l] = rl * rl + EPSF;
        }
        __syncthreads();
    }
}

// ---------------- K6: scalar scans per (b,h): running max / sum-exp / cumsum(gates) ----------------
// grid = B*H, block = 32 (one wave, wave32 shuffles)
__global__ void scan_kernel(const float* __restrict__ sim, const float* __restrict__ gates,
                            float* __restrict__ m_s, float* __restrict__ s_s, float* __restrict__ g_s) {
    int bh = blockIdx.x, lane = threadIdx.x;
    float cm = -1e30f, cs = 0.f, cg = 0.f;
    for (int c0 = 0; c0 < LL; c0 += 32) {
        long i = (long)bh * LL + c0 + lane;
        float m = sim[i], s = 1.f, g = gates[i];
#pragma unroll
        for (int off = 1; off < 32; off <<= 1) {
            float pm = __shfl_up(m, off);
            float ps = __shfl_up(s, off);
            float pg = __shfl_up(g, off);
            if (lane >= off) {
                float mn = fmaxf(pm, m);
                s = ps * expf(pm - mn) + s * expf(m - mn);
                m = mn;
                g += pg;
            }
        }
        float mn = fmaxf(cm, m);
        float st = cs * expf(cm - mn) + s * expf(m - mn);
        float gt = cg + g;
        m_s[i] = mn; s_s[i] = st; g_s[i] = gt;
        cm = __shfl(mn, 31); cs = __shfl(st, 31); cg = __shfl(gt, 31);
    }
}

// ---------------- K7: causal linear attention via WMMA ----------------
// ctxt_raw[l,e] = sum_{m<=l} gate_m * (q_l . v_t_m) * k_t[m,e]
// ctxt[l,e] = ctxt_raw[l,e] * (1 + silu(exp(sim_l - m_s_l)/(s_s_l+EPS))) / (g_s_l+EPS) * kv_scale[hd*4096]
// grid = (B*H, L/64), block = 128 (4 waves, 16 q-rows each)
__global__ void attn_kernel(const unsigned short* __restrict__ qbf,
                            const unsigned short* __restrict__ ktT,
                            const unsigned short* __restrict__ vtbf,
                            const float* __restrict__ sim, const float* __restrict__ m_s,
                            const float* __restrict__ s_s, const float* __restrict__ g_s,
                            const float* __restrict__ gates, const float* __restrict__ kv_scale,
                            unsigned short* __restrict__ ctxt_bf) {
    int bh = blockIdx.x, b = bh >> 3, hd = bh & 7;
    int wave = threadIdx.x >> 5, lane = threadIdx.x & 31;
    int half = lane >> 4, nn = lane & 15;
    int lblk = blockIdx.y * 64;
    int l0 = lblk + wave * 16;

    __shared__ __align__(16) unsigned short s_lds[4][16][32];
    __shared__ float c_lds[64];

    if (threadIdx.x < 64) {
        int l = lblk + threadIdx.x;
        long idx = (long)bh * LL + l;
        float w = expf(sim[idx] - m_s[idx]) / (s_s[idx] + EPSF);
        float sil = w / (1.f + expf(-w));
        c_lds[threadIdx.x] = (1.f + sil) / (g_s[idx] + EPSF) * kv_scale[hd * 4096];
    }
    __syncthreads();

    const unsigned short* qrow = qbf + ((long)(b * LL + l0)) * DD + hd * 64;
    v16bf a0 = load_frag_rm(qrow, DD, lane);        // d = 0..31
    v16bf a1 = load_frag_rm(qrow + 32, DD, lane);   // d = 32..63

    const unsigned short* vtb  = vtbf + (long)b * LL * DD + hd * 64;
    const unsigned short* ktTb = ktT + (long)bh * 64 * LL;   // rows = e, cols = m (contiguous)

    v8f acc[4];
#pragma unroll
    for (int et = 0; et < 4; ++et) acc[et] = (v8f){0.f, 0.f, 0.f, 0.f, 0.f, 0.f, 0.f, 0.f};

    for (int m0 = 0; m0 <= l0 + 15; m0 += 32) {
        // --- S = q @ v_t^T on this 16 x 32 m-chunk, masked + gated -> LDS (bf16) ---
#pragma unroll
        for (int sub = 0; sub < 2; ++sub) {
            int mt = m0 + sub * 16;
            v16bf b0 = load_frag_rm_clamp(vtb, DD, mt, LL - 1, 0, lane);
            v16bf b1 = load_frag_rm_clamp(vtb, DD, mt, LL - 1, 32, lane);
            v8f s = {0.f, 0.f, 0.f, 0.f, 0.f, 0.f, 0.f, 0.f};
            s = wmma_bf16(a0, b0, s);
            s = wmma_bf16(a1, b1, s);
            int m = mt + nn;
            float g = (m < LL) ? gates[(long)bh * LL + m] : 0.f;
#pragma unroll
            for (int r = 0; r < 8; ++r) {
                int l = l0 + r + half * 8;
                float sv = (m <= l) ? s[r] * g : 0.f;
                s_lds[wave][r + half * 8][sub * 16 + nn] = f2bf(sv);
            }
        }
        // --- ctxt += S @ k_t (contraction over m; B from pre-transposed k_t, contiguous) ---
        v16bf sA = load_frag_rm(&s_lds[wave][0][0], 32, lane);
#pragma unroll
        for (int et = 0; et < 4; ++et) {
            v16bf bk = load_frag_rm(ktTb + (long)(et * 16) * LL + m0, LL, lane);
            acc[et] = wmma_bf16(sA, bk, acc[et]);
        }
    }

#pragma unroll
    for (int et = 0; et < 4; ++et) {
#pragma unroll
        for (int r = 0; r < 8; ++r) {
            int lrow = wave * 16 + r + half * 8;
            int l = lblk + lrow;
            float v = acc[et][r] * c_lds[lrow];
            ctxt_bf[((long)(b * LL + l)) * DD + hd * 64 + et * 16 + nn] = f2bf(v);
        }
    }
}

// ---------------- launch ----------------
extern "C" void kernel_launch(void* const* d_in, const int* in_sizes, int n_in,
                              void* d_out, int out_size, void* d_ws, size_t ws_size,
                              hipStream_t stream) {
    const float* x        = (const float*)d_in[0];
    const float* Wq       = (const float*)d_in[1];
    const float* bq       = (const float*)d_in[2];
    const float* Wk       = (const float*)d_in[3];
    const float* bk       = (const float*)d_in[4];
    const float* Wv       = (const float*)d_in[5];
    const float* bv       = (const float*)d_in[6];
    const float* Wo       = (const float*)d_in[7];
    const float* bo       = (const float*)d_in[8];
    const float* Wg       = (const float*)d_in[9];
    const float* bg       = (const float*)d_in[10];
    const float* Wtd      = (const float*)d_in[11];
    const float* btd      = (const float*)d_in[12];
    const float* qk_scale = (const float*)d_in[13];
    const float* kv_scale = (const float*)d_in[14];
    const float* sf       = (const float*)d_in[15];

    const int M = BB * LL;              // 2048
    const long ND = (long)M * DD;       // 1,048,576

    char* cur = (char*)d_ws;
    auto alloc = [&](size_t bytes) { void* p = (void*)cur; cur += (bytes + 255) & ~(size_t)255; return p; };

    unsigned short* xbf   = (unsigned short*)alloc(ND * 2);
    unsigned short* Wqt   = (unsigned short*)alloc(DD * DD * 2);
    unsigned short* Wkt   = (unsigned short*)alloc(DD * DD * 2);
    unsigned short* Wvt   = (unsigned short*)alloc(DD * DD * 2);
    unsigned short* Wot   = (unsigned short*)alloc(DD * DD * 2);
    float*          qf    = (float*)alloc(ND * 4);
    float*          kf    = (float*)alloc(ND * 4);
    float*          vf    = (float*)alloc(ND * 4);
    unsigned short* qbf   = (unsigned short*)alloc(ND * 2);
    float*          fp    = (float*)alloc((long)LL * DD * 4);
    float*          kt    = (float*)alloc(ND * 4);
    float*          vt    = (float*)alloc(ND * 4);
    unsigned short* ktT   = (unsigned short*)alloc(ND * 2 + 512);  // pad: tail tiles over-read <=62B
    unsigned short* vtbf  = (unsigned short*)alloc(ND * 2);
    float*          simb  = (float*)alloc((long)BB * HH * LL * 4);
    float*          gates = (float*)alloc((long)BB * HH * LL * 4);
    float*          msb   = (float*)alloc((long)BB * HH * LL * 4);
    float*          ssb   = (float*)alloc((long)BB * HH * LL * 4);
    float*          gsb   = (float*)alloc((long)BB * HH * LL * 4);
    unsigned short* ctxtb = (unsigned short*)alloc(ND * 2);
    (void)ws_size; (void)n_in; (void)in_sizes; (void)out_size;

    // K0: conversions / weight transposes
    cvt_f32_to_bf16<<<2048, 256, 0, stream>>>(x, xbf, (int)ND);
    transpose_w_bf16<<<(DD * DD + 255) / 256, 256, 0, stream>>>(Wq, Wqt);
    transpose_w_bf16<<<(DD * DD + 255) / 256, 256, 0, stream>>>(Wk, Wkt);
    transpose_w_bf16<<<(DD * DD + 255) / 256, 256, 0, stream>>>(Wv, Wvt);
    transpose_w_bf16<<<(DD * DD + 255) / 256, 256, 0, stream>>>(Wo, Wot);

    // K1: spectral filter projection
    fproj_kernel<<<(LL * DD + 255) / 256, 256, 0, stream>>>(sf, Wtd, btd, fp);

    // K2: q,k,v projections (WMMA bf16, async-staged A tiles)
    dim3 ggrid(M / 16, DD / 64);
    gemm_bf16_kernel<<<ggrid, 128, 0, stream>>>(xbf, Wqt, bq, qf, qbf, M, DD);
    gemm_bf16_kernel<<<ggrid, 128, 0, stream>>>(xbf, Wkt, bk, kf, (unsigned short*)nullptr, M, DD);
    gemm_bf16_kernel<<<ggrid, 128, 0, stream>>>(xbf, Wvt, bv, vf, (unsigned short*)nullptr, M, DD);

    // K3: sim + normalize
    simnorm_kernel<<<M, DD, 0, stream>>>(qf, kf, vf, qk_scale, simb);

    // K4: causal convolutions (emits f32 + bf16 row-major v_t + transposed bf16 k_t)
    conv_kernel<<<dim3(BB * HH, LL / 4), dim3(64, 4), 0, stream>>>(kf, vf, fp, kt, vt, ktT, vtbf);

    // K5: gates
    gate_kernel<<<dim3(BB * HH, LL / 64), 64, 0, stream>>>(kt, vt, Wg, bg, kv_scale, gates);

    // K6: scalar prefix scans
    scan_kernel<<<BB * HH, 32, 0, stream>>>(simb, gates, msb, ssb, gsb);

    // K7: WMMA causal linear attention -> ctxt (bf16)
    attn_kernel<<<dim3(BB * HH, LL / 64), 128, 0, stream>>>(qbf, ktT, vtbf, simb, msb, ssb, gsb,
                                                            gates, kv_scale, ctxtb);

    // K8: output projection straight into d_out (f32)
    gemm_bf16_kernel<<<ggrid, 128, 0, stream>>>(ctxtb, Wot, bo, (float*)d_out,
                                                (unsigned short*)nullptr, M, DD);
}
